// MultiHeadAttention_8280696946761
// MI455X (gfx1250) — matrix operand order repaired and checked by reading, not compiled, not run on hardware
//
#include <hip/hip_runtime.h>
#include <type_traits>

typedef _Float16 f16;
typedef __attribute__((ext_vector_type(16))) _Float16 v16h;
typedef __attribute__((ext_vector_type(8)))  _Float16 v8h;
typedef __attribute__((ext_vector_type(8)))  float    v8f;
typedef unsigned int uint32;

#define BATCH 8
#define SEQ   1024
#define HID   1024
#define NHEAD 16
#define DH    64

static __device__ __forceinline__ v8f zero8() { v8f z = {}; return z; }

// ---- gfx1250 async global->LDS copy (ASYNCcnt-tracked), GVS addressing ----
// global addr = SGPR64(base) + VGPR32(offset); LDS dest = per-lane VGPR addr.
static __device__ __forceinline__ void async_copy_b128(uint32 lds_byte_addr,
                                                       const void* sbase,
                                                       uint32 voff_bytes) {
  asm volatile("global_load_async_to_lds_b128 %0, %1, %2"
               :
               : "v"(lds_byte_addr), "v"(voff_bytes), "s"(sbase)
               : "memory");
}
static __device__ __forceinline__ void wait_async0() {
  asm volatile("s_wait_asynccnt 0" ::: "memory");
}
// LDS byte offset of a __shared__ object: low 32 bits of its flat address.
static __device__ __forceinline__ uint32 lds_off(const void* p) {
  return (uint32)(uintptr_t)p;
}

// K offset of dword j within a 16x32 f16 A-fragment:
// VGPR0..7 hold contiguous K pairs {0,1}..{14,15};
// lanes 16..31 are shifted by +16 in K.
static __device__ __forceinline__ int kpair_off(int j, int half) {
  return 2 * j + half * 16;
}

static __device__ __forceinline__ v16h load_afrag(const f16* base_row, int half) {
  v16h a;
  uint32* ap = reinterpret_cast<uint32*>(&a);
#pragma unroll
  for (int j = 0; j < 8; ++j)
    ap[j] = *reinterpret_cast<const uint32*>(base_row + kpair_off(j, half));
  return a;
}

// 32x16 B-fragment: lane n holds column n; 16 contiguous K halves
// (lanes 16..31 take K=16..31). `base` includes col*stride + half*16.
static __device__ __forceinline__ v16h load_bfrag(const f16* base) {
  v16h b;
  v8h* bp = reinterpret_cast<v8h*>(&b);
  bp[0] = *reinterpret_cast<const v8h*>(base);
  bp[1] = *reinterpret_cast<const v8h*>(base + 8);
  return b;
}

// ---------------------------------------------------------------------------
// C[M,N] = A[M,K] @ W[N,K]^T + bias   (NT GEMM, f32 accumulate via f16 WMMA)
// Block: 128 threads = 4 waves; 128x128 block tile; 64x64 wave tile
// (4x4 WMMA subtiles -> 8 fragment loads feed 16 WMMAs per K-step).
// AT=f16 A-tiles stream via GLOBAL_LOAD_ASYNC_TO_LDS_B128.
// ---------------------------------------------------------------------------
template <typename AT, typename OT>
__global__ __launch_bounds__(128) void gemm_nt_kernel(
    const AT* __restrict__ A, const float* __restrict__ W,
    const float* __restrict__ bias, OT* __restrict__ C,
    int M, int N, int K) {
  constexpr int LDT = 40;  // padded LDS stride (halves); 80B rows keep 16B align
  __shared__ __align__(16) f16 As[128 * LDT];
  __shared__ __align__(16) f16 Bs[128 * LDT];

  const int tid  = threadIdx.x;  // 0..127
  const int lane = tid & 31;
  const int wid  = tid >> 5;     // 0..3
  const int wm   = wid >> 1;     // 0..1
  const int wn   = wid & 1;      // 0..1
  const int half = lane >> 4;
  const int l16  = lane & 15;

  const int bm = blockIdx.y * 128;
  const int bn = blockIdx.x * 128;

  v8f acc[4][4];
#pragma unroll
  for (int i = 0; i < 4; ++i)
#pragma unroll
    for (int j = 0; j < 4; ++j) acc[i][j] = zero8();

  for (int kb = 0; kb < K; kb += 32) {
    // A tile: thread tid owns tile row tid (32 k-elements).
    if constexpr (std::is_same<AT, f16>::value) {
      const uint32 goff = (uint32)(((size_t)(bm + tid) * K + kb) * sizeof(f16));
      const uint32 loff = lds_off(As + tid * LDT);
#pragma unroll
      for (int s = 0; s < 4; ++s)  // 4 x 16B = 64B row
        async_copy_b128(loff + s * 16, A, goff + s * 16);
    } else {
      const AT* src = A + (size_t)(bm + tid) * K + kb;
      f16* dst = As + tid * LDT;
#pragma unroll
      for (int i = 0; i < 32; ++i) dst[i] = (f16)(float)src[i];
      __builtin_prefetch(src + 32, 0, 1);
    }
    // W tile (always fp32 -> f16 convert in VGPRs).
    {
      const float* src = W + (size_t)(bn + tid) * K + kb;
      f16* dst = Bs + tid * LDT;
#pragma unroll
      for (int i = 0; i < 32; ++i) dst[i] = (f16)src[i];
      __builtin_prefetch(src + 32, 0, 1);
    }
    if constexpr (std::is_same<AT, f16>::value) wait_async0();
    __syncthreads();

    v16h af[4], bf[4];
#pragma unroll
    for (int i = 0; i < 4; ++i)
      af[i] = load_afrag(As + (wm * 64 + i * 16 + l16) * LDT, half);
#pragma unroll
    for (int j = 0; j < 4; ++j)
      bf[j] = load_bfrag(Bs + (wn * 64 + j * 16 + l16) * LDT + half * 16);

#pragma unroll
    for (int i = 0; i < 4; ++i)
#pragma unroll
      for (int j = 0; j < 4; ++j)
        acc[i][j] = __builtin_amdgcn_wmma_f32_16x16x32_f16(
            false, af[i], false, bf[j], (short)0, acc[i][j], false, false);

    __syncthreads();
  }

  // Epilogue: C/D layout — lane holds col n=l16, VGPR r holds row r+8*half.
#pragma unroll
  for (int j = 0; j < 4; ++j) {
    const int col = bn + wn * 64 + j * 16 + l16;
    const float bv = bias[col];
#pragma unroll
    for (int i = 0; i < 4; ++i)
#pragma unroll
      for (int r = 0; r < 8; ++r) {
        const int row = bm + wm * 64 + i * 16 + r + 8 * half;
        C[(size_t)row * N + col] = (OT)(acc[i][j][r] + bv);
      }
  }
}

// ---------------------------------------------------------------------------
// Flash-style attention with post-softmax masking per the reference.
// Block = 128 threads (4 waves); one block per (b, h, 64 q rows); each wave
// owns 16 q rows. 32-key tiles: K streamed async to LDS, V transposed on
// store so PV B-fragments are contiguous ds_load_b128.
// ---------------------------------------------------------------------------
__global__ __launch_bounds__(128) void attn_kernel(
    const f16* __restrict__ qp, const f16* __restrict__ kp,
    const f16* __restrict__ vp, const float* __restrict__ qmask,
    const float* __restrict__ kmask, f16* __restrict__ ctx) {
  constexpr int LKS = 72;  // Ks stride: [key][d], 64 -> 72 pad (144B rows)
  constexpr int LVT = 40;  // Vt stride: [d][key], 32 -> 40 pad (80B rows)
  constexpr int LPS = 40;  // Ps stride: [m][key]
  __shared__ __align__(16) f16 Ks[32 * LKS];
  __shared__ __align__(16) f16 Vt[64 * LVT];
  __shared__ __align__(16) f16 Ps[4][16 * LPS];
  __shared__ float kms[32];

  const int tid  = threadIdx.x;
  const int lane = tid & 31;
  const int w    = tid >> 5;  // wave 0..3
  const int half = lane >> 4;
  const int l16  = lane & 15;

  const int qt = blockIdx.x;  // q tile of 64
  const int h  = blockIdx.y;  // head
  const int b  = blockIdx.z;  // batch

  const int q0 = qt * 64 + w * 16;  // first q row of this wave

  // Preload q A-fragments (two K-steps over DH=64).
  v16h qf[2];
  {
    const f16* qrow = qp + (size_t)(b * SEQ + q0 + l16) * HID + h * DH;
#pragma unroll
    for (int t = 0; t < 2; ++t) qf[t] = load_afrag(qrow + t * 32, half);
  }

  v8f accd[4];
#pragma unroll
  for (int dt = 0; dt < 4; ++dt) accd[dt] = zero8();
  float mrun[8], lrun[8];
#pragma unroll
  for (int r = 0; r < 8; ++r) { mrun[r] = -1e30f; lrun[r] = 0.f; }

  const int key  = tid >> 2;        // 0..31
  const int dseg = (tid & 3) * 16;  // 0,16,32,48

  for (int kt = 0; kt < SEQ; kt += 32) {
    __syncthreads();  // everyone done reading previous Ks/Vt
    {
      // K tile: async global->LDS (f16, no conversion needed).
      const uint32 kgoff = (uint32)(((size_t)(b * SEQ + kt + key) * HID +
                                     h * DH + dseg) * sizeof(f16));
      const uint32 kloff = lds_off(Ks + key * LKS + dseg);
      async_copy_b128(kloff, kp, kgoff);
      async_copy_b128(kloff + 16, kp, kgoff + 16);

      // V tile: load through VGPRs, transpose on store.
      const f16* vsrc = vp + (size_t)(b * SEQ + kt + key) * HID + h * DH + dseg;
      v8h v0 = *reinterpret_cast<const v8h*>(vsrc);
      v8h v1 = *reinterpret_cast<const v8h*>(vsrc + 8);
      __builtin_prefetch(vsrc + 32 * HID, 0, 1);
#pragma unroll
      for (int i = 0; i < 8; ++i) {
        Vt[(dseg + i) * LVT + key]     = v0[i];
        Vt[(dseg + 8 + i) * LVT + key] = v1[i];
      }
      if (tid < 32) kms[tid] = kmask[b * SEQ + kt + tid];
    }
    wait_async0();
    __syncthreads();

    // Scores: two 16-key subtiles, K-dim = DH (two 32-wide WMMA steps).
    v8f sa[2];
#pragma unroll
    for (int j = 0; j < 2; ++j) {
      sa[j] = zero8();
#pragma unroll
      for (int t = 0; t < 2; ++t) {
        v16h bf = load_bfrag(Ks + (j * 16 + l16) * LKS + t * 32 + half * 16);
        sa[j] = __builtin_amdgcn_wmma_f32_16x16x32_f16(
            false, qf[t], false, bf, (short)0, sa[j], false, false);
      }
    }
    const bool km0 = kms[l16] > 0.f;
    const bool km1 = kms[16 + l16] > 0.f;

    float s0[8], s1[8], mc[8];
#pragma unroll
    for (int r = 0; r < 8; ++r) {
      s0[r] = km0 ? sa[0][r] * 8.0f : -1e30f;  // scale = sqrt(DH) per reference
      s1[r] = km1 ? sa[1][r] * 8.0f : -1e30f;
      mc[r] = fmaxf(s0[r], s1[r]);
    }
#pragma unroll
    for (int d = 1; d < 16; d <<= 1)
#pragma unroll
      for (int r = 0; r < 8; ++r)
        mc[r] = fmaxf(mc[r], __shfl_xor(mc[r], d, 32));

    float alpha[8], psum[8];
#pragma unroll
    for (int r = 0; r < 8; ++r) {
      const float mnew = fmaxf(mrun[r], mc[r]);
      alpha[r] = __expf(mrun[r] - mnew);
      s0[r] = km0 ? __expf(s0[r] - mnew) : 0.f;
      s1[r] = km1 ? __expf(s1[r] - mnew) : 0.f;
      psum[r] = s0[r] + s1[r];
      mrun[r] = mnew;
    }
#pragma unroll
    for (int d = 1; d < 16; d <<= 1)
#pragma unroll
      for (int r = 0; r < 8; ++r) psum[r] += __shfl_xor(psum[r], d, 32);
#pragma unroll
    for (int r = 0; r < 8; ++r) lrun[r] = alpha[r] * lrun[r] + psum[r];

#pragma unroll
    for (int dt = 0; dt < 4; ++dt)
#pragma unroll
      for (int r = 0; r < 8; ++r) accd[dt][r] *= alpha[r];

    // P (C/D layout) -> LDS row-major so it can be re-read as an A-fragment.
#pragma unroll
    for (int r = 0; r < 8; ++r) {
      Ps[w][(r + 8 * half) * LPS + l16]      = (f16)s0[r];
      Ps[w][(r + 8 * half) * LPS + 16 + l16] = (f16)s1[r];
    }
    __syncthreads();

    // ctx += P(16x32) x V(32x64): 4 WMMAs over d-subtiles.
    v16h pf = load_afrag(&Ps[w][l16 * LPS], half);
#pragma unroll
    for (int dt = 0; dt < 4; ++dt) {
      v16h vf = load_bfrag(Vt + (dt * 16 + l16) * LVT + half * 16);
      accd[dt] = __builtin_amdgcn_wmma_f32_16x16x32_f16(
          false, pf, false, vf, (short)0, accd[dt], false, false);
    }
  }

  // Finalize: renormalize by kept-probability sum; zero Q-masked rows.
  float rl[8], qm[8];
#pragma unroll
  for (int r = 0; r < 8; ++r) {
    rl[r] = 1.0f / (lrun[r] + 1e-6f);
    qm[r] = qmask[b * SEQ + q0 + r + 8 * half];
  }
#pragma unroll
  for (int dt = 0; dt < 4; ++dt)
#pragma unroll
    for (int r = 0; r < 8; ++r) {
      const int row = b * SEQ + q0 + r + 8 * half;
      ctx[(size_t)row * HID + h * DH + dt * 16 + l16] =
          (f16)(accd[dt][r] * rl[r] * qm[r]);
    }
}

// ---------------------------------------------------------------------------
extern "C" void kernel_launch(void* const* d_in, const int* in_sizes, int n_in,
                              void* d_out, int out_size, void* d_ws,
                              size_t ws_size, hipStream_t stream) {
  (void)in_sizes; (void)n_in; (void)out_size; (void)ws_size;
  const float* Q  = (const float*)d_in[0];
  const float* K  = (const float*)d_in[1];
  const float* V  = (const float*)d_in[2];
  const float* Qm = (const float*)d_in[3];
  const float* Km = (const float*)d_in[4];
  const float* Wq = (const float*)d_in[5];
  const float* bq = (const float*)d_in[6];
  const float* Wk = (const float*)d_in[7];
  const float* bk = (const float*)d_in[8];
  const float* Wv = (const float*)d_in[9];
  const float* bv = (const float*)d_in[10];
  const float* Wo = (const float*)d_in[11];
  const float* bo = (const float*)d_in[12];

  const size_t tsz = (size_t)BATCH * SEQ * HID;  // 8M elements
  f16* qp = (f16*)d_ws;  // 16 MB each; 64 MB total workspace use,
  f16* kp = qp + tsz;    // L2-resident on MI455X (192 MB L2).
  f16* vp = kp + tsz;
  f16* cx = vp + tsz;

  const int M = BATCH * SEQ;       // 8192
  dim3 ggrid(HID / 128, M / 128);  // (8, 64)

  gemm_nt_kernel<float, f16><<<ggrid, 128, 0, stream>>>(Q, Wq, bq, qp, M, HID, HID);
  gemm_nt_kernel<float, f16><<<ggrid, 128, 0, stream>>>(K, Wk, bk, kp, M, HID, HID);
  gemm_nt_kernel<float, f16><<<ggrid, 128, 0, stream>>>(V, Wv, bv, vp, M, HID, HID);

  attn_kernel<<<dim3(SEQ / 64, NHEAD, BATCH), 128, 0, stream>>>(qp, kp, vp, Qm, Km, cx);

  gemm_nt_kernel<f16, float><<<ggrid, 128, 0, stream>>>(cx, Wo, bo, (float*)d_out, M, HID, HID);
}
